// UniversalGraphStack_25211458028166
// MI455X (gfx1250) — compile-verified, matrix operand order checked by reading, not run on hardware
//
#include <hip/hip_runtime.h>
#include <hip/hip_bf16.h>
#include <stdint.h>

// ---------------------------------------------------------------------------
// GCN stack for MI455X (gfx1250, wave32).
//   layer: h = x @ W^T  (WMMA bf16->f32, K fully unrolled)
//          agg[dst] += h[src]*norm  (edge-parallel atomic f32 scatter)
//          v = agg + b + h*dis^2    (self loop) + BN stats (LDS reduce)
//          out = relu((v-mean)*rsqrt(var+eps)*g+be) (+ residual layers 1,2)
// ---------------------------------------------------------------------------

typedef __attribute__((ext_vector_type(16))) __bf16 v16bf;
typedef __attribute__((ext_vector_type(8)))  float  v8f;

#define HID 64
#define BN_BLOCKS 512

// ---------------- utility ----------------
__global__ void zero_kernel(float* __restrict__ p, int n) {
    int i = blockIdx.x * blockDim.x + threadIdx.x;
    if (i < n) p[i] = 0.0f;
}

// ---------------- degree / norm ----------------
__global__ void deg_kernel(const long long* __restrict__ dst, float* __restrict__ deg, int E) {
    int e = blockIdx.x * blockDim.x + threadIdx.x;
    if (e < E) atomicAdd(&deg[(int)dst[e]], 1.0f);
}

__global__ void dis_kernel(float* __restrict__ dis, int N) {
    int i = blockIdx.x * blockDim.x + threadIdx.x;
    if (i < N) dis[i] = rsqrtf(dis[i] + 1.0f);   // +1 self loop; always >= 1
}

__global__ void norm_kernel(const long long* __restrict__ src, const long long* __restrict__ dst,
                            const float* __restrict__ dis, float* __restrict__ norm, int E) {
    int e = blockIdx.x * blockDim.x + threadIdx.x;
    if (e < E) norm[e] = dis[(int)src[e]] * dis[(int)dst[e]];
}

// ---------------- GEMM: H[N x 64] = X[N x K] @ W[64 x K]^T  (bf16 WMMA) ----
// block = 256 threads = 8 waves; wave -> one 16x16 output tile.
// wave>>2 selects node tile (0..1), wave&3 selects 16-channel tile (0..3).
// grid.x = N/32.  K templated -> loop fully unrolled, straight-line wmma.
template <int K>
__global__ void gemm_wmma(const float* __restrict__ X, const float* __restrict__ W,
                          float* __restrict__ H) {
    const int tid  = threadIdx.x;
    const int wave = tid >> 5;
    const int lane = tid & 31;
    const int nb = blockIdx.x * 32 + (wave >> 2) * 16;   // node base
    const int cb = (wave & 3) * 16;                       // output-channel base
    const int m     = lane & 15;          // A row / B col owned by this lane
    const int khalf = (lane >> 4) << 3;   // 0 (lanes 0-15) or 8 (lanes 16-31)

    v8f c = {0.f, 0.f, 0.f, 0.f, 0.f, 0.f, 0.f, 0.f};
    const float* xrow = X + (size_t)(nb + m) * K;
    const float* wrow = W + (size_t)(cb + m) * K;

#pragma unroll
    for (int k0 = 0; k0 < K; k0 += 32) {
        // Each lane owns two runs of 8 consecutive K values:
        //   halves 0..7  : K = k0 + khalf + {0..7}
        //   halves 8..15 : K = k0 + 16 + khalf + {0..7}
        // -> 2x float4 loads per run per matrix (b128).
        const float4* xa = reinterpret_cast<const float4*>(xrow + k0 + khalf);
        const float4* wb = reinterpret_cast<const float4*>(wrow + k0 + khalf);
        float4 x0 = xa[0], x1 = xa[1];          // K = khalf+0..7
        float4 x2 = xa[4], x3 = xa[5];          // K = 16+khalf+0..7   (offset 16 floats)
        float4 w0 = wb[0], w1 = wb[1];
        float4 w2 = wb[4], w3 = wb[5];

        v16bf a, b;
        a[0]=(__bf16)x0.x; a[1]=(__bf16)x0.y; a[2]=(__bf16)x0.z; a[3]=(__bf16)x0.w;
        a[4]=(__bf16)x1.x; a[5]=(__bf16)x1.y; a[6]=(__bf16)x1.z; a[7]=(__bf16)x1.w;
        a[8]=(__bf16)x2.x; a[9]=(__bf16)x2.y; a[10]=(__bf16)x2.z; a[11]=(__bf16)x2.w;
        a[12]=(__bf16)x3.x; a[13]=(__bf16)x3.y; a[14]=(__bf16)x3.z; a[15]=(__bf16)x3.w;
        b[0]=(__bf16)w0.x; b[1]=(__bf16)w0.y; b[2]=(__bf16)w0.z; b[3]=(__bf16)w0.w;
        b[4]=(__bf16)w1.x; b[5]=(__bf16)w1.y; b[6]=(__bf16)w1.z; b[7]=(__bf16)w1.w;
        b[8]=(__bf16)w2.x; b[9]=(__bf16)w2.y; b[10]=(__bf16)w2.z; b[11]=(__bf16)w2.w;
        b[12]=(__bf16)w3.x; b[13]=(__bf16)w3.y; b[14]=(__bf16)w3.z; b[15]=(__bf16)w3.w;

        c = __builtin_amdgcn_wmma_f32_16x16x32_bf16(
                /*neg_a=*/false, a, /*neg_b=*/false, b,
                /*c_mod=*/(short)0, c, /*reuse_a=*/false, /*reuse_b=*/false);
    }

    // D layout: VGPR r -> M = r (+8 for lanes 16..31), N = lane&15
    const int mrow = (lane >> 4) << 3;
#pragma unroll
    for (int r = 0; r < 8; ++r)
        H[(size_t)(nb + mrow + r) * HID + cb + (lane & 15)] = c[r];
}

// ---------------- edge scatter: agg[dst] += h[src] * norm --------------
// one wave per edge; lane j handles channels {2j, 2j+1}
__global__ void aggregate_kernel(const float* __restrict__ h, const float* __restrict__ norm,
                                 const long long* __restrict__ src, const long long* __restrict__ dst,
                                 float* __restrict__ agg, int E) {
    int t = blockIdx.x * blockDim.x + threadIdx.x;
    int e = t >> 5;
    if (e >= E) return;
    int lane = t & 31;
    int s = (int)src[e];
    int d = (int)dst[e];
    float nm = norm[e];
    float2 v = *reinterpret_cast<const float2*>(h + (size_t)s * HID + 2 * lane);
    atomicAdd(agg + (size_t)d * HID + 2 * lane,     v.x * nm);
    atomicAdd(agg + (size_t)d * HID + 2 * lane + 1, v.y * nm);
}

// ---------------- bias + self loop + BN statistics ---------------------
// block = 256: c = tid&63 (channel), yr = tid>>6 (row group 0..3)
__global__ void post_agg_stats(float* __restrict__ agg, const float* __restrict__ h,
                               const float* __restrict__ dis, const float* __restrict__ bias,
                               float* __restrict__ stats, int N) {
    const int c  = threadIdx.x & 63;
    const int yr = threadIdx.x >> 6;
    float s = 0.f, ss = 0.f;
    for (int n = blockIdx.x * 4 + yr; n < N; n += gridDim.x * 4) {
        float dv = dis[n];
        size_t idx = (size_t)n * HID + c;
        float v = agg[idx] + bias[c] + h[idx] * dv * dv;
        agg[idx] = v;
        s += v;  ss += v * v;
    }
    __shared__ float sh[512];
    sh[threadIdx.x]       = s;
    sh[256 + threadIdx.x] = ss;
    __syncthreads();
    if (yr == 0) {
        float ts  = sh[c] + sh[64 + c] + sh[128 + c] + sh[192 + c];
        float tss = sh[256 + c] + sh[320 + c] + sh[384 + c] + sh[448 + c];
        atomicAdd(&stats[c], ts);
        atomicAdd(&stats[64 + c], tss);
    }
}

// ---------------- BN apply + relu + residual ---------------------------
__global__ void bn_finalize(const float* __restrict__ agg, const float* __restrict__ stats,
                            const float* __restrict__ g, const float* __restrict__ be,
                            const float* __restrict__ resid, float* __restrict__ out,
                            int total, int residual, float invN) {
    int i = blockIdx.x * blockDim.x + threadIdx.x;
    if (i >= total) return;
    int c = i & 63;
    float mean = stats[c] * invN;
    float var  = stats[64 + c] * invN - mean * mean;
    float v = (agg[i] - mean) * rsqrtf(var + 1e-5f) * g[c] + be[c];
    v = fmaxf(v, 0.0f);
    if (residual) v += resid[i];
    out[i] = v;
}

// ---------------------------------------------------------------------------
extern "C" void kernel_launch(void* const* d_in, const int* in_sizes, int n_in,
                              void* d_out, int out_size, void* d_ws, size_t ws_size,
                              hipStream_t stream) {
    const int IN_DIM = 128;
    const int N = in_sizes[0] / IN_DIM;      // 100000
    const int E = in_sizes[1] / 2;           // 1600000

    const float*      x  = (const float*)d_in[0];
    const long long*  ei = (const long long*)d_in[1];
    const long long*  src = ei;
    const long long*  dst = ei + E;
    const float* W[3]  = {(const float*)d_in[2],  (const float*)d_in[6],  (const float*)d_in[10]};
    const float* b[3]  = {(const float*)d_in[3],  (const float*)d_in[7],  (const float*)d_in[11]};
    const float* g[3]  = {(const float*)d_in[4],  (const float*)d_in[8],  (const float*)d_in[12]};
    const float* be[3] = {(const float*)d_in[5],  (const float*)d_in[9],  (const float*)d_in[13]};

    // ---- workspace carve-up (256B aligned) ----
    auto align256 = [](size_t v) { return (v + 255) & ~(size_t)255; };
    char* ws = (char*)d_ws;
    float* dis  = (float*)ws;  ws += align256((size_t)N * 4);
    float* norm = (float*)ws;  ws += align256((size_t)E * 4);
    float* h    = (float*)ws;  ws += align256((size_t)N * HID * 4);
    float* agg  = (float*)ws;  ws += align256((size_t)N * HID * 4);
    float* xcur = (float*)ws;  ws += align256((size_t)N * HID * 4);
    float* stats = (float*)ws; ws += align256(128 * 4);

    const int TB = 256;
    const int gN   = (N + TB - 1) / TB;
    const int gE   = (E + TB - 1) / TB;
    const int gNH  = (N * HID + TB - 1) / TB;
    const int gAgg = ((E * 32) + TB - 1) / TB;
    const int gGemm = N / 32;                 // N is a multiple of 32

    // degree -> dis -> per-edge norm (reused by all 3 layers)
    zero_kernel<<<gN, TB, 0, stream>>>(dis, N);
    deg_kernel<<<gE, TB, 0, stream>>>(dst, dis, E);
    dis_kernel<<<gN, TB, 0, stream>>>(dis, N);
    norm_kernel<<<gE, TB, 0, stream>>>(src, dst, dis, norm, E);

    const float invN = 1.0f / (float)N;
    for (int l = 0; l < 3; ++l) {
        const float* xin = (l == 0) ? x : xcur;
        const int residual = (l == 0) ? 0 : 1;
        float* outp = (l == 2) ? (float*)d_out : xcur;

        zero_kernel<<<gNH, TB, 0, stream>>>(agg, N * HID);
        zero_kernel<<<1, 128, 0, stream>>>(stats, 128);

        if (l == 0) gemm_wmma<128><<<gGemm, TB, 0, stream>>>(xin, W[l], h);
        else        gemm_wmma<64> <<<gGemm, TB, 0, stream>>>(xin, W[l], h);

        aggregate_kernel<<<gAgg, TB, 0, stream>>>(h, norm, src, dst, agg, E);
        post_agg_stats<<<BN_BLOCKS, TB, 0, stream>>>(agg, h, dis, b[l], stats, N);
        bn_finalize<<<gNH, TB, 0, stream>>>(agg, stats, g[l], be[l],
                                            xin, outp, N * HID, residual, invN);
    }
}
